// MoELayerOptimized_57569741635631
// MI455X (gfx1250) — matrix operand order repaired and checked by reading, not compiled, not run on hardware
//
#include <hip/hip_runtime.h>
#include <hip/hip_bf16.h>
#include <math.h>

// ---------------- problem dims (fixed by reference) ----------------
#define T_TOK   8192          // B*S = 4*2048 tokens
#define HDIM    1024
#define FDIM    4096
#define NEXP    8
#define TS      32            // token-expert pairs per MoE tile (M tile)
#define LDSROW  (FDIM + 8)    // padded LDS row stride (bf16 elems): 8208 B, 16B-aligned, bank-conflict free

typedef __attribute__((ext_vector_type(8)))  float  v8f;
typedef __attribute__((ext_vector_type(16))) __bf16 v16bf;
typedef unsigned short ushort_t;

// round-to-nearest-even f32 -> bf16 (raw bits)
__device__ __forceinline__ ushort_t f2bf(float f) {
  union { float f; unsigned u; } v; v.f = f;
  unsigned r = v.u + 0x7fffu + ((v.u >> 16) & 1u);
  return (ushort_t)(r >> 16);
}

// build a v16bf fragment from two 16-byte chunks (global or LDS)
__device__ __forceinline__ v16bf frag16(const void* p0, const void* p1) {
  union { v16bf v; uint4 q[2]; } u;
  u.q[0] = *(const uint4*)p0;
  u.q[1] = *(const uint4*)p1;
  return u.v;
}

// ---------------- init: zero out, pair lists, counters ----------------
__global__ void k_init(float* __restrict__ out, int* __restrict__ pairTok,
                       float* __restrict__ pairW, int* __restrict__ cnt) {
  int i = blockIdx.x * blockDim.x + threadIdx.x;
  if (i < T_TOK * HDIM) out[i] = 0.0f;
  if (i < NEXP * T_TOK) { pairTok[i] = 0; pairW[i] = 0.0f; }
  if (i < NEXP) cnt[i] = 0;
}

// ---------------- x -> bf16 ----------------
__global__ void k_cvt_x(const float* __restrict__ x, ushort_t* __restrict__ xb) {
  int i = blockIdx.x * blockDim.x + threadIdx.x;      // one float4 per thread
  float4 v = ((const float4*)x)[i];
  union { ushort_t s[4]; uint2 u; } p;
  p.s[0] = f2bf(v.x); p.s[1] = f2bf(v.y); p.s[2] = f2bf(v.z); p.s[3] = f2bf(v.w);
  *(uint2*)(xb + (size_t)i * 4) = p.u;
}

// ---------------- W[e][K][N] f32 -> Wt[e][N][K] bf16 (tiled transpose) ----------------
__global__ void k_cvt_wT(const float* __restrict__ src, ushort_t* __restrict__ dst,
                         int K, int N) {
  __shared__ float tile[32][33];
  const int e  = blockIdx.z;
  const int k0 = blockIdx.x * 32;
  const int n0 = blockIdx.y * 32;
  const int tx = threadIdx.x, ty = threadIdx.y;
  const float* s = src + (size_t)e * K * N;
  ushort_t*    d = dst + (size_t)e * K * N;
  for (int r = ty; r < 32; r += 8)
    tile[r][tx] = s[(size_t)(k0 + r) * N + n0 + tx];
  __syncthreads();
  for (int r = ty; r < 32; r += 8)
    d[(size_t)(n0 + r) * K + k0 + tx] = f2bf(tile[tx][r]);
}

// ---------------- router: softmax top-2 + renorm, append to expert segments ----------------
__global__ void k_router(const float* __restrict__ x, const float* __restrict__ Wr,
                         const float* __restrict__ br,
                         int* __restrict__ pairTok, float* __restrict__ pairW,
                         int* __restrict__ cnt) {
  const int wave = threadIdx.x >> 5;
  const int lane = threadIdx.x & 31;
  const int t = blockIdx.x * 8 + wave;          // one wave32 per token
  const float* xr = x + (size_t)t * HDIM;
  float l[8] = {0,0,0,0,0,0,0,0};
  for (int h = lane; h < HDIM; h += 32) {
    const float xv = xr[h];
    const float4 w0 = ((const float4*)(Wr + (size_t)h * NEXP))[0];
    const float4 w1 = ((const float4*)(Wr + (size_t)h * NEXP))[1];
    l[0] += xv * w0.x; l[1] += xv * w0.y; l[2] += xv * w0.z; l[3] += xv * w0.w;
    l[4] += xv * w1.x; l[5] += xv * w1.y; l[6] += xv * w1.z; l[7] += xv * w1.w;
  }
  #pragma unroll
  for (int e = 0; e < 8; ++e)
    #pragma unroll
    for (int off = 16; off > 0; off >>= 1)
      l[e] += __shfl_xor(l[e], off, 32);
  if (lane == 0) {
    float p[8];
    float mx = -3.4e38f;
    for (int e = 0; e < 8; ++e) { l[e] += br[e]; if (l[e] > mx) mx = l[e]; }
    for (int e = 0; e < 8; ++e) p[e] = expf(l[e] - mx);   // softmax denom cancels in renorm
    int i1 = 0;
    for (int e = 1; e < 8; ++e) if (p[e] > p[i1]) i1 = e; // ties -> lowest idx (matches top_k)
    int i2 = (i1 == 0) ? 1 : 0;
    for (int e = 0; e < 8; ++e) if (e != i1 && p[e] > p[i2]) i2 = e;
    const float s = p[i1] + p[i2];
    int pos = atomicAdd(&cnt[i1], 1);
    pairTok[i1 * T_TOK + pos] = t; pairW[i1 * T_TOK + pos] = p[i1] / s;
    pos = atomicAdd(&cnt[i2], 1);
    pairTok[i2 * T_TOK + pos] = t; pairW[i2 * T_TOK + pos] = p[i2] / s;
  }
}

// ---------------- fused expert FFN: gelu(x@W1+b1)@W2+b2, weighted scatter-add ----------------
// block = 256 threads (8 waves), tile = 32 pairs; h[32][F] bf16 lives in 263 KB of LDS.
__global__ void __launch_bounds__(256)
k_moe_ffn(const ushort_t* __restrict__ xb,
          const ushort_t* __restrict__ w1t,    // [E][F][H] bf16
          const ushort_t* __restrict__ w2t,    // [E][H][F] bf16
          const float* __restrict__ b1, const float* __restrict__ b2,
          const int* __restrict__ pairTok, const float* __restrict__ pairW,
          const int* __restrict__ cnt, float* __restrict__ out) {
  extern __shared__ char smem[];
  ushort_t* hT  = (ushort_t*)smem;                               // [TS][LDSROW]
  int*     sTok = (int*)(smem + (size_t)TS * LDSROW * 2);
  float*   sW   = (float*)(sTok + TS);

  const int e    = blockIdx.x >> 8;     // 256 tiles per expert segment
  const int m0   = (blockIdx.x & 255) * TS;
  if (m0 >= cnt[e]) return;             // uniform early-out past this expert's pair count

  const int tid = threadIdx.x;
  if (tid < TS) {                       // pad slots were pre-zeroed: tok 0 / weight 0
    sTok[tid] = pairTok[e * T_TOK + m0 + tid];
    sW[tid]   = pairW [e * T_TOK + m0 + tid];
  }
  __syncthreads();

  const int wave = tid >> 5;
  const int lane = tid & 31;
  const int l15  = lane & 15;
  const int hi   = lane >> 4;
  const int kbA  = hi * 8;    // 16-bit A layout: chunks at K=kbA and K=kbA+16
  const int kbB  = hi * 8;    // 16-bit B layout: chunks at K=kbB and K=kbB+16

  // ---- phase 1: hT = gelu(x[toks] @ W1[e] + b1[e]) ----
  const ushort_t* aRow0 = xb + (size_t)sTok[l15]      * HDIM;
  const ushort_t* aRow1 = xb + (size_t)sTok[l15 + 16] * HDIM;
  const ushort_t* w1e = w1t + (size_t)e * FDIM * HDIM;

  for (int nc = 0; nc < 8; ++nc) {                 // wave owns 512 F-cols, 64 at a time
    const int nBase = wave * 512 + nc * 64;
    v8f acc[2][4] = {};
    const ushort_t* bCol[4];
    #pragma unroll
    for (int nf = 0; nf < 4; ++nf)
      bCol[nf] = w1e + (size_t)(nBase + nf * 16 + l15) * HDIM;
    for (int k0 = 0; k0 < HDIM; k0 += 32) {
      v16bf a0 = frag16(aRow0 + k0 + kbA, aRow0 + k0 + kbA + 16);
      v16bf a1 = frag16(aRow1 + k0 + kbA, aRow1 + k0 + kbA + 16);
      #pragma unroll
      for (int nf = 0; nf < 4; ++nf) {
        v16bf b = frag16(bCol[nf] + k0 + kbB, bCol[nf] + k0 + kbB + 16);
        acc[0][nf] = __builtin_amdgcn_wmma_f32_16x16x32_bf16(false, a0, false, b, (short)0, acc[0][nf], false, false);
        acc[1][nf] = __builtin_amdgcn_wmma_f32_16x16x32_bf16(false, a1, false, b, (short)0, acc[1][nf], false, false);
      }
    }
    #pragma unroll
    for (int mf = 0; mf < 2; ++mf)
      #pragma unroll
      for (int nf = 0; nf < 4; ++nf) {
        const int col = nBase + nf * 16 + l15;
        const float bias = b1[e * FDIM + col];
        #pragma unroll
        for (int v = 0; v < 8; ++v) {
          const int row = mf * 16 + hi * 8 + v;    // D layout: lanes 0-15 -> M=v, 16-31 -> M=8+v
          float z = acc[mf][nf][v] + bias;
          float g = 0.5f * z * (1.0f + erff(z * 0.70710678118f));  // exact GELU
          hT[row * LDSROW + col] = f2bf(g);
        }
      }
  }
  __syncthreads();

  // ---- phase 2: y = hT @ W2[e] + b2[e]; out[tok] += w * y ----
  const ushort_t* w2e = w2t + (size_t)e * HDIM * FDIM;
  const ushort_t* aR0 = hT + (size_t)l15 * LDSROW;
  const ushort_t* aR1 = hT + (size_t)(l15 + 16) * LDSROW;
  for (int nc = 0; nc < 2; ++nc) {                 // wave owns 128 H-cols, 64 at a time
    const int nBase = wave * 128 + nc * 64;
    v8f acc[2][4] = {};
    const ushort_t* bCol[4];
    #pragma unroll
    for (int nf = 0; nf < 4; ++nf)
      bCol[nf] = w2e + (size_t)(nBase + nf * 16 + l15) * FDIM;
    for (int k0 = 0; k0 < FDIM; k0 += 32) {
      v16bf a0 = frag16(aR0 + k0 + kbA, aR0 + k0 + kbA + 16);   // ds_load_b128 x2
      v16bf a1 = frag16(aR1 + k0 + kbA, aR1 + k0 + kbA + 16);
      #pragma unroll
      for (int nf = 0; nf < 4; ++nf) {
        v16bf b = frag16(bCol[nf] + k0 + kbB, bCol[nf] + k0 + kbB + 16);
        acc[0][nf] = __builtin_amdgcn_wmma_f32_16x16x32_bf16(false, a0, false, b, (short)0, acc[0][nf], false, false);
        acc[1][nf] = __builtin_amdgcn_wmma_f32_16x16x32_bf16(false, a1, false, b, (short)0, acc[1][nf], false, false);
      }
    }
    #pragma unroll
    for (int mf = 0; mf < 2; ++mf)
      #pragma unroll
      for (int nf = 0; nf < 4; ++nf) {
        const int col = nBase + nf * 16 + l15;
        const float bias = b2[e * HDIM + col];
        #pragma unroll
        for (int v = 0; v < 8; ++v) {
          const int row = mf * 16 + hi * 8 + v;
          const float w = sW[row];
          if (w != 0.0f) {   // exactly 2 contributions per out element -> deterministic
            atomicAdd(&out[(size_t)sTok[row] * HDIM + col], (acc[mf][nf][v] + bias) * w);
          }
        }
      }
  }
}

// ---------------- launcher ----------------
extern "C" void kernel_launch(void* const* d_in, const int* in_sizes, int n_in,
                              void* d_out, int out_size, void* d_ws, size_t ws_size,
                              hipStream_t stream) {
  const float* x  = (const float*)d_in[0];   // [4,2048,1024]
  const float* Wr = (const float*)d_in[1];   // [1024,8]
  const float* br = (const float*)d_in[2];   // [8]
  const float* W1 = (const float*)d_in[3];   // [8,1024,4096]
  const float* b1 = (const float*)d_in[4];   // [8,4096]
  const float* W2 = (const float*)d_in[5];   // [8,4096,1024]
  const float* b2 = (const float*)d_in[6];   // [8,1024]
  float* out = (float*)d_out;
  char* ws = (char*)d_ws;

  // workspace layout (needs ~151.5 MB)
  const size_t SZ_XB = (size_t)T_TOK * HDIM * 2;          // 16 MB bf16 x
  const size_t SZ_W  = (size_t)NEXP * HDIM * FDIM * 2;    // 64 MB each bf16 weight
  ushort_t* xb      = (ushort_t*)(ws);
  ushort_t* w1t     = (ushort_t*)(ws + SZ_XB);
  ushort_t* w2t     = (ushort_t*)(ws + SZ_XB + SZ_W);
  int*      pairTok = (int*)  (ws + SZ_XB + 2 * SZ_W);
  float*    pairW   = (float*)(ws + SZ_XB + 2 * SZ_W + (size_t)NEXP * T_TOK * 4);
  int*      cnt     = (int*)  (ws + SZ_XB + 2 * SZ_W + (size_t)NEXP * T_TOK * 8);

  k_init<<<(T_TOK * HDIM + 255) / 256, 256, 0, stream>>>(out, pairTok, pairW, cnt);
  k_cvt_x<<<(T_TOK * HDIM / 4) / 256, 256, 0, stream>>>(x, xb);
  k_cvt_wT<<<dim3(HDIM / 32, FDIM / 32, NEXP), dim3(32, 8), 0, stream>>>(W1, w1t, HDIM, FDIM);
  k_cvt_wT<<<dim3(FDIM / 32, HDIM / 32, NEXP), dim3(32, 8), 0, stream>>>(W2, w2t, FDIM, HDIM);
  k_router<<<T_TOK / 8, 256, 0, stream>>>(x, Wr, br, pairTok, pairW, cnt);

  const int SMEM = TS * LDSROW * 2 + TS * 8;   // 262,912 B of the WGP's 320 KB LDS
  k_moe_ffn<<<NEXP * (T_TOK / TS), 256, SMEM, stream>>>(
      xb, w1t, w2t, b1, b2, pairTok, pairW, cnt, out);
}
